// GPT_50302656971081
// MI455X (gfx1250) — matrix operand order repaired and checked by reading, not compiled, not run on hardware
//
#include <hip/hip_runtime.h>

// ---------------- model constants ----------------
#define BB 2
#define TT 1024
#define DD 1024
#define HH 16
#define DH 64
#define MM 4096
#define VV 50257
#define LL 4
#define BT (BB * TT)   // 2048 token rows

typedef __bf16 v16bf __attribute__((ext_vector_type(16)));
typedef float  v8f   __attribute__((ext_vector_type(8)));
typedef unsigned short us8  __attribute__((ext_vector_type(8)));
typedef unsigned int   u32x8 __attribute__((ext_vector_type(8)));

// ---------------- helpers ----------------
__device__ __forceinline__ unsigned short f2bf_bits(float f) {
  unsigned u = __builtin_bit_cast(unsigned, f);
  u += 0x7FFFu + ((u >> 16) & 1u);          // round-to-nearest-even
  return (unsigned short)(u >> 16);
}
__device__ __forceinline__ float bfbits2f(unsigned short h) {
  unsigned u = ((unsigned)h) << 16;
  return __builtin_bit_cast(float, u);
}
// load a lane's 32B fragment (16 bf16) from LDS -> two ds_load_b128
__device__ __forceinline__ v16bf ld_frag(const unsigned short* p) {
  u32x8 r = *reinterpret_cast<const u32x8*>(p);
  return __builtin_bit_cast(v16bf, r);
}
__device__ __forceinline__ v16bf frag_from_f(const float* f) {
  us8 lo, hi;
#pragma unroll
  for (int j = 0; j < 8; ++j) { lo[j] = f2bf_bits(f[j]); hi[j] = f2bf_bits(f[8 + j]); }
  u32x8 r;
#pragma unroll
  for (int j = 0; j < 4; ++j) {
    r[j]     = (unsigned)lo[2 * j] | ((unsigned)lo[2 * j + 1] << 16);
    r[4 + j] = (unsigned)hi[2 * j] | ((unsigned)hi[2 * j + 1] << 16);
  }
  return __builtin_bit_cast(v16bf, r);
}
// A-matrix 16x32 fragment addressing (inverse maps)
//   forward: lane_hi=k[3], e = (k[4]<<3)|(k&7)
__device__ __forceinline__ int a_e(int k)  { return ((k >> 4) << 3) | (k & 7); }
__device__ __forceinline__ int a_hi(int k) { return (k >> 3) & 1; }
__device__ __forceinline__ float gelu_exact(float x) {
  return 0.5f * x * (1.0f + erff(x * 0.70710678118654752f));
}

// ---------------- embedding ----------------
__global__ void gpt_embed_kernel(const int* __restrict__ x,
                                 const float* __restrict__ tok,
                                 const float* __restrict__ pos,
                                 float* __restrict__ out) {
  int gid = blockIdx.x * blockDim.x + threadIdx.x;
  int i4  = gid * 4;
  if (i4 >= BT * DD) return;
  int row = i4 / DD, t = row % TT, d = i4 % DD;
  int tk  = x[row];
  const float4 te = *reinterpret_cast<const float4*>(&tok[(size_t)tk * DD + d]);
  const float4 pe = *reinterpret_cast<const float4*>(&pos[(size_t)t  * DD + d]);
  float4 o{te.x + pe.x, te.y + pe.y, te.z + pe.z, te.w + pe.w};
  *reinterpret_cast<float4*>(&out[i4]) = o;
}

// ---------------- layernorm ----------------
__global__ void gpt_ln_kernel(const float* __restrict__ in,
                              const float* __restrict__ g,
                              const float* __restrict__ b,
                              float* __restrict__ out) {
  __shared__ float s1[256];
  __shared__ float s2[256];
  const int row = blockIdx.x, tid = threadIdx.x;
  const float* r = in + (size_t)row * DD;
  float v[4], sum = 0.f, sq = 0.f;
#pragma unroll
  for (int i = 0; i < 4; ++i) {
    v[i] = r[tid + i * 256];
    sum += v[i]; sq += v[i] * v[i];
  }
  s1[tid] = sum; s2[tid] = sq;
  __syncthreads();
  for (int s = 128; s > 0; s >>= 1) {
    if (tid < s) { s1[tid] += s1[tid + s]; s2[tid] += s2[tid + s]; }
    __syncthreads();
  }
  const float mu  = s1[0] * (1.0f / DD);
  const float var = s2[0] * (1.0f / DD) - mu * mu;
  const float rs  = rsqrtf(var + 1e-5f);
  float* o = out + (size_t)row * DD;
#pragma unroll
  for (int i = 0; i < 4; ++i) {
    int d = tid + i * 256;
    o[d] = (v[i] - mu) * rs * g[d] + b[d];
  }
}

// ---------------- bf16-WMMA GEMM, fragment-major LDS staging ----------------
// block 256 thr = 8 waves; tile 64(M) x 128(N); K-step 32.
// sA: 4 subtiles (16x32) x 32 lanes x 16 bf16 (fragment order), 4 KB
// sB: 8 subtiles (32x16) x 32 lanes x 16 bf16 (fragment order), 8 KB
__global__ void gpt_gemm_bf16_kernel(const float* __restrict__ A,
                                     const float* __restrict__ W,
                                     const float* __restrict__ bias,
                                     const float* __restrict__ resid,
                                     float* __restrict__ C,
                                     int Mrows, int N, int K, int act) {
  __shared__ unsigned short sA[4 * 32 * 16];
  __shared__ unsigned short sB[8 * 32 * 16];
  const int tid  = threadIdx.x;
  const int lane = tid & 31;
  const int w    = tid >> 5;
  const int wm   = w >> 1;
  const int wn   = w & 1;
  const int row0 = blockIdx.y * 64;
  const int col0 = blockIdx.x * 128;

  v8f acc[4];
#pragma unroll
  for (int nt = 0; nt < 4; ++nt) acc[nt] = v8f{0.f,0.f,0.f,0.f,0.f,0.f,0.f,0.f};

  // A staging geometry: thread -> (row, 8-wide k chunk); one b128 LDS store
  const int ar = tid >> 2;
  const int akc = (tid & 3) * 8;
  const int a_dst = (ar >> 4) * 512 + ((a_hi(akc) * 16 + (ar & 15)) << 4) + a_e(akc);

  for (int k0 = 0; k0 < K; k0 += 32) {
    // ---- stage A (64x32): two float4 global loads, one b128 LDS store ----
    {
      const float* src = &A[(size_t)(row0 + ar) * K + (k0 + akc)];
      float4 f0 = *reinterpret_cast<const float4*>(src);
      float4 f1 = *reinterpret_cast<const float4*>(src + 4);
      us8 pk;
      pk[0]=f2bf_bits(f0.x); pk[1]=f2bf_bits(f0.y); pk[2]=f2bf_bits(f0.z); pk[3]=f2bf_bits(f0.w);
      pk[4]=f2bf_bits(f1.x); pk[5]=f2bf_bits(f1.y); pk[6]=f2bf_bits(f1.z); pk[7]=f2bf_bits(f1.w);
      *reinterpret_cast<us8*>(&sA[a_dst]) = pk;
    }
    // ---- stage B (32x128): float4 global loads, fragment-order scatter ----
#pragma unroll
    for (int i = 0; i < 4; ++i) {
      int idx4 = tid + i * 256;           // 1024 float4 groups
      int r = idx4 >> 5;                  // k row 0..31
      int cb = (idx4 & 31) * 4;           // col 0..124
      float fv[4];
      int nbase = col0 + cb;
      if (nbase + 3 < N) {
        float4 t = *reinterpret_cast<const float4*>(&W[(size_t)(k0 + r) * N + nbase]);
        fv[0] = t.x; fv[1] = t.y; fv[2] = t.z; fv[3] = t.w;
      } else {
#pragma unroll
        for (int j = 0; j < 4; ++j)
          fv[j] = (nbase + j < N) ? W[(size_t)(k0 + r) * N + nbase + j] : 0.f;
      }
#pragma unroll
      for (int j = 0; j < 4; ++j) {
        int c = cb + j;
        int dst = (((c >> 4) * 32 + (a_hi(r) << 4) + (c & 15)) << 4) + a_e(r);
        sB[dst] = f2bf_bits(fv[j]);
      }
    }
    if (k0 + 32 < K) {
      __builtin_prefetch(&A[(size_t)(row0 + ar) * K + (k0 + 32 + akc)], 0, 0);
      __builtin_prefetch(&W[(size_t)(k0 + 32 + (tid >> 3)) * N + col0], 0, 0);
    }
    __syncthreads();

    v16bf afr = ld_frag(&sA[(wm * 512) + (lane << 4)]);
#pragma unroll
    for (int nt = 0; nt < 4; ++nt) {
      v16bf bfr = ld_frag(&sB[(((wn * 4 + nt) * 32 + lane) << 4)]);
      acc[nt] = __builtin_amdgcn_wmma_f32_16x16x32_bf16(
          false, afr, false, bfr, (short)0, acc[nt], false, false);
    }
    __syncthreads();
  }

  // epilogue
#pragma unroll
  for (int nt = 0; nt < 4; ++nt) {
#pragma unroll
    for (int r = 0; r < 8; ++r) {
      int gr = row0 + wm * 16 + ((lane >> 4) << 3) + r;
      int gc = col0 + wn * 64 + nt * 16 + (lane & 15);
      if (gc < N) {
        float vv = acc[nt][r] + bias[gc];
        if (act == 1) vv = gelu_exact(vv);
        if (resid) vv += resid[(size_t)gr * N + gc];
        C[(size_t)gr * N + gc] = vv;
      }
    }
  }
}

// ---------------- fused attention ----------------
// sPF holds 16x1024 scores/probs in A-fragment-major order (bf16, 32 KB):
//   element (row q, key j): ks=j>>5, krel=j&31,
//   idx = ((ks*32 + a_hi(krel)*16 + q) << 4) + a_e(krel)
__device__ __forceinline__ int pf_idx(int q, int j) {
  int ks = j >> 5, kr = j & 31;
  return ((ks * 32 + a_hi(kr) * 16 + q) << 4) + a_e(kr);
}

__global__ void gpt_attn_kernel(const float* __restrict__ qkv,
                                float* __restrict__ inner) {
  __shared__ unsigned short sPF[32 * 32 * 16];  // 32 KB
  __shared__ float sO[16][DH];                  // 4 KB

  const int tid  = threadIdx.x;
  const int lane = tid & 31;
  const int w    = tid >> 5;
  const int qt   = blockIdx.x & 63;
  const int h    = (blockIdx.x >> 6) & (HH - 1);
  const int b    = blockIdx.x >> 10;
  const int q0   = qt * 16;

  const size_t third = (size_t)BT * DD;
  const float* Qp = qkv + (size_t)b * TT * DD + h * DH;
  const float* Kp = Qp + third;
  const float* Vp = Qp + 2 * third;

#pragma unroll
  for (int i = 0; i < 4; ++i) {               // zero partial-output buffer
    int e = tid + i * 256;
    sO[e >> 6][e & 63] = 0.f;
  }

  // ---- phase 1: S = Q K^T / 8 ----
  const int tq = q0 + (lane & 15);
  float qb[16];
  v16bf aq0, aq1;
  {
    const int hi = lane >> 4;
#pragma unroll
    for (int e = 0; e < 16; ++e) {            // k for element e: hi*16 + e (contiguous)
      int v = e >> 1;
      int k = (hi << 4) + e;
      qb[e] = Qp[(size_t)tq * DD + k];
    }
    aq0 = frag_from_f(qb);
#pragma unroll
    for (int e = 0; e < 16; ++e) {
      int v = e >> 1;
      int k = (hi << 4) + e;
      qb[e] = Qp[(size_t)tq * DD + 32 + k];
    }
    aq1 = frag_from_f(qb);
  }
  for (int kt = w; kt < TT / 16; kt += 8) {
    const int tk = kt * 16 + (lane & 15);
    const int dlo = (lane >> 4) << 4;
    float kf[16];
#pragma unroll
    for (int c = 0; c < 4; ++c) {
      float4 t = *reinterpret_cast<const float4*>(&Kp[(size_t)tk * DD + dlo + c * 4]);
      kf[c*4+0]=t.x; kf[c*4+1]=t.y; kf[c*4+2]=t.z; kf[c*4+3]=t.w;
    }
    v16bf bk0 = frag_from_f(kf);
#pragma unroll
    for (int c = 0; c < 4; ++c) {
      float4 t = *reinterpret_cast<const float4*>(&Kp[(size_t)tk * DD + 32 + dlo + c * 4]);
      kf[c*4+0]=t.x; kf[c*4+1]=t.y; kf[c*4+2]=t.z; kf[c*4+3]=t.w;
    }
    v16bf bk1 = frag_from_f(kf);
    v8f s = v8f{0.f,0.f,0.f,0.f,0.f,0.f,0.f,0.f};
    s = __builtin_amdgcn_wmma_f32_16x16x32_bf16(false, aq0, false, bk0, (short)0, s, false, false);
    s = __builtin_amdgcn_wmma_f32_16x16x32_bf16(false, aq1, false, bk1, (short)0, s, false, false);
#pragma unroll
    for (int r = 0; r < 8; ++r) {
      int qrow = ((lane >> 4) << 3) + r;
      sPF[pf_idx(qrow, kt * 16 + (lane & 15))] = f2bf_bits(s[r] * 0.125f);
    }
  }
  __syncthreads();

  // ---- phase 2: softmax over keys; wave w owns rows 2w, 2w+1 ----
#pragma unroll
  for (int rr = 0; rr < 2; ++rr) {
    const int row = w * 2 + rr;
    float m = -1e30f;
    for (int j = lane; j < TT; j += 32) m = fmaxf(m, bfbits2f(sPF[pf_idx(row, j)]));
#pragma unroll
    for (int off = 16; off > 0; off >>= 1) m = fmaxf(m, __shfl_xor(m, off, 32));
    float sum = 0.f;
    for (int j = lane; j < TT; j += 32) {
      int ix = pf_idx(row, j);
      float e = __expf(bfbits2f(sPF[ix]) - m);
      sum += e;
      sPF[ix] = f2bf_bits(e);
    }
#pragma unroll
    for (int off = 16; off > 0; off >>= 1) sum += __shfl_xor(sum, off, 32);
    const float inv = 1.0f / sum;
    for (int j = lane; j < TT; j += 32) {
      int ix = pf_idx(row, j);
      sPF[ix] = f2bf_bits(bfbits2f(sPF[ix]) * inv);
    }
  }
  __syncthreads();

  // ---- phase 3: O = P V, K split over wave halves, ds_add_f32 combine ----
  const int nt = w & 3;
  const int kh = w >> 2;
  v8f o = v8f{0.f,0.f,0.f,0.f,0.f,0.f,0.f,0.f};
  for (int ks = kh * 16; ks < kh * 16 + 16; ++ks) {
    v16bf ap = ld_frag(&sPF[(ks * 32 + lane) << 4]);
    const int kb = ks * 32;
    float vf[16];
#pragma unroll
    for (int e = 0; e < 16; ++e)
      vf[e] = Vp[(size_t)(kb + ((e >> 3) << 4) + ((lane >> 4) << 3) + (e & 7)) * DD + nt * 16 + (lane & 15)];
    v16bf bv = frag_from_f(vf);
    o = __builtin_amdgcn_wmma_f32_16x16x32_bf16(false, ap, false, bv, (short)0, o, false, false);
  }
#pragma unroll
  for (int r = 0; r < 8; ++r)
    atomicAdd(&sO[((lane >> 4) << 3) + r][nt * 16 + (lane & 15)], o[r]);
  __syncthreads();

#pragma unroll
  for (int i = 0; i < 4; ++i) {
    int e = tid + i * 256;
    int q = e >> 6, d = e & 63;
    inner[(size_t)b * TT * DD + (size_t)(q0 + q) * DD + h * DH + d] = sO[q][d];
  }
}

// ---------------- host orchestration ----------------
extern "C" void kernel_launch(void* const* d_in, const int* in_sizes, int n_in,
                              void* d_out, int out_size, void* d_ws, size_t ws_size,
                              hipStream_t stream) {
  (void)in_sizes; (void)n_in; (void)out_size; (void)ws_size;
  const int*   x       = (const int*)  d_in[0];
  const float* tok_emb = (const float*)d_in[1];
  const float* pos_emb = (const float*)d_in[2];
  const float* ln1_g   = (const float*)d_in[3];
  const float* ln1_b   = (const float*)d_in[4];
  const float* qkv_w   = (const float*)d_in[5];
  const float* qkv_b   = (const float*)d_in[6];
  const float* out_w   = (const float*)d_in[7];
  const float* out_b   = (const float*)d_in[8];
  const float* ln2_g   = (const float*)d_in[9];
  const float* ln2_b   = (const float*)d_in[10];
  const float* wi_w    = (const float*)d_in[11];
  const float* wi_b    = (const float*)d_in[12];
  const float* wo_w    = (const float*)d_in[13];
  const float* wo_b    = (const float*)d_in[14];
  const float* ln_g    = (const float*)d_in[15];
  const float* ln_b    = (const float*)d_in[16];
  const float* head_w  = (const float*)d_in[17];
  const float* head_b  = (const float*)d_in[18];
  float* logits = (float*)d_out;

  float* W    = (float*)d_ws;
  float* hbuf = W;
  float* lnb  = W + (size_t)1 * BT * DD;
  float* qkvb = W + (size_t)2 * BT * DD;
  float* innb = W + (size_t)5 * BT * DD;
  float* hidb = W + (size_t)6 * BT * DD;

  const dim3 blk(256);

  gpt_embed_kernel<<<(BT * DD / 4 + 255) / 256, blk, 0, stream>>>(x, tok_emb, pos_emb, hbuf);

  for (int l = 0; l < LL; ++l) {
    const float* qw  = qkv_w + (size_t)l * DD * 3 * DD;
    const float* qb  = qkv_b + (size_t)l * 3 * DD;
    const float* ow  = out_w + (size_t)l * DD * DD;
    const float* ob  = out_b + (size_t)l * DD;
    const float* wiw = wi_w  + (size_t)l * DD * MM;
    const float* wib = wi_b  + (size_t)l * MM;
    const float* wow = wo_w  + (size_t)l * MM * DD;
    const float* wob = wo_b  + (size_t)l * DD;

    gpt_ln_kernel<<<BT, blk, 0, stream>>>(hbuf, ln1_g + l * DD, ln1_b + l * DD, lnb);
    gpt_gemm_bf16_kernel<<<dim3(3 * DD / 128, BT / 64), blk, 0, stream>>>(
        lnb, qw, qb, nullptr, qkvb, BT, 3 * DD, DD, 0);
    gpt_attn_kernel<<<BB * HH * (TT / 16), blk, 0, stream>>>(qkvb, innb);
    gpt_gemm_bf16_kernel<<<dim3(DD / 128, BT / 64), blk, 0, stream>>>(
        innb, ow, ob, hbuf, hbuf, BT, DD, DD, 0);

    gpt_ln_kernel<<<BT, blk, 0, stream>>>(hbuf, ln2_g + l * DD, ln2_b + l * DD, lnb);
    gpt_gemm_bf16_kernel<<<dim3(MM / 128, BT / 64), blk, 0, stream>>>(
        lnb, wiw, wib, nullptr, hidb, BT, MM, DD, 1 /*gelu*/);
    gpt_gemm_bf16_kernel<<<dim3(DD / 128, BT / 64), blk, 0, stream>>>(
        hidb, wow, wob, hbuf, hbuf, BT, DD, MM, 0);
  }

  gpt_ln_kernel<<<BT, blk, 0, stream>>>(hbuf, ln_g, ln_b, lnb);
  gpt_gemm_bf16_kernel<<<dim3((VV + 127) / 128, BT / 64), blk, 0, stream>>>(
      lnb, head_w, head_b, nullptr, logits, BT, VV, DD, 0);
}
